// DogDetector_36713380446430
// MI455X (gfx1250) — matrix operand order repaired and checked by reading, not compiled, not run on hardware
//
#include <hip/hip_runtime.h>

typedef unsigned short u16;
typedef unsigned int   u32;

typedef __bf16 v16bf __attribute__((ext_vector_type(16)));
typedef float  v8f   __attribute__((ext_vector_type(8)));
typedef u32    u32x4 __attribute__((ext_vector_type(4)));
typedef int    i32x8 __attribute__((ext_vector_type(8)));
typedef int    i32x4 __attribute__((ext_vector_type(4)));

#if __has_builtin(__builtin_amdgcn_tensor_load_to_lds) && __has_builtin(__builtin_amdgcn_s_wait_tensorcnt)
#define USE_TDM 1
#else
#define USE_TDM 0
#endif

#if __has_builtin(__builtin_amdgcn_global_load_async_to_lds_b128) && __has_builtin(__builtin_amdgcn_s_wait_asynccnt)
#define USE_ASYNC 1
#else
#define USE_ASYNC 0
#endif

// b128 async-copy payload type, as required by the builtin's prototype
// (param0: int vector_size(16) in AS1/global; param1: same in AS3/LDS).
typedef int vi4 __attribute__((vector_size(16)));
typedef __attribute__((address_space(1))) vi4* gvi4p;
typedef __attribute__((address_space(3))) vi4* lvi4p;

#define BATCH 128
#define C512  512
#define C256  256
#define H14   14
#define H7    7
#define M14   (BATCH * H14 * H14)   // 25088
#define M7    (BATCH * H7 * H7)     // 6272
#define K33   (9 * C256)            // 2304
#define NANCH 441
#define NMSN  512

__device__ __forceinline__ u16 f2bf(float f) {
  u32 u = __float_as_uint(f);
  u32 r = (u + 0x7FFFu + ((u >> 16) & 1u)) >> 16;   // round-to-nearest-even
  return (u16)r;
}
__device__ __forceinline__ float bf2f(u16 h) {
  return __uint_as_float(((u32)h) << 16);
}

union Frag { uint4 q[2]; v16bf v; };

// ---------------------------------------------------------------------------
// 0) zero-fill helper (padded-lane source for async im2col loads)
// ---------------------------------------------------------------------------
__global__ void zero_fill(u16* __restrict__ p) { p[threadIdx.x] = 0; }

// ---------------------------------------------------------------------------
// 1) NCHW fp32 features -> [B*H*W, C] bf16 rows
// ---------------------------------------------------------------------------
__global__ void cvt_features(const float* __restrict__ in, u16* __restrict__ out) {
  size_t i = (size_t)blockIdx.x * blockDim.x + threadIdx.x;
  size_t total = (size_t)M14 * C512;
  if (i >= total) return;
  int c = (int)(i & (C512 - 1));
  size_t mp = i >> 9;                 // row index b*196+p
  int p = (int)(mp % (H14 * H14));
  int b = (int)(mp / (H14 * H14));
  out[i] = f2bf(in[((size_t)b * C512 + c) * (H14 * H14) + p]);
}

// ---------------------------------------------------------------------------
// 2) OIHW fp32 weights -> transposed packed bf16 [O, taps*I], k = t*I + ci
// ---------------------------------------------------------------------------
__global__ void pack_wT(const float* __restrict__ w, u16* __restrict__ wT,
                        int O, int I, int KH) {
  int taps = KH * KH;
  size_t K = (size_t)taps * I;
  size_t total = (size_t)O * K;
  size_t i = (size_t)blockIdx.x * blockDim.x + threadIdx.x;
  if (i >= total) return;
  int k = (int)(i % K);
  int o = (int)(i / K);
  int t = k / I, ci = k % I;
  int ky = t / KH, kx = t % KH;
  wT[i] = f2bf(w[(((size_t)o * I + ci) * KH + ky) * KH + kx]);
}

// ---------------------------------------------------------------------------
// 3) fold conv-bias + eval BN into per-channel scale/shift
// ---------------------------------------------------------------------------
__global__ void make_ss(const float* __restrict__ bias, const float* __restrict__ gamma,
                        const float* __restrict__ beta, const float* __restrict__ mean,
                        const float* __restrict__ var,
                        float* __restrict__ scale, float* __restrict__ shift) {
  int c = threadIdx.x;
  float s = gamma[c] * rsqrtf(var[c] + 1e-5f);
  scale[c] = s;
  shift[c] = (bias[c] - mean[c]) * s + beta[c];
}

// ---------------------------------------------------------------------------
// 4) implicit-GEMM conv (1x1 or 3x3 pad1), bf16 WMMA, fused BN(+ReLU)
//    act : [B*H*W, Cin] bf16 rows ; wT : [256, Ktot] bf16 ; out : [B*H*W, 256]
//    block = 256 threads (8 waves); block tile 128(M) x 128(N); wave 32x64.
//    A tile: GLOBAL_LOAD_ASYNC_TO_LDS (ASYNCcnt), double buffered; padded
//            lanes read a zeroed 32-byte workspace region instead.
//    B tile: Tensor Data Mover DMA (wave0 issues, TENSORcnt), double buffered.
//    One barrier per K-chunk: each wave waits its ASYNCcnt, wave0 also waits
//    TENSORcnt, then s_barrier publishes both transfers.
// ---------------------------------------------------------------------------
__global__ __launch_bounds__(256)
void gemm_bn_wmma(const u16* __restrict__ act, const u16* __restrict__ wT,
                  const float* __restrict__ scale, const float* __restrict__ shift,
                  u16* __restrict__ out, const u16* __restrict__ zeroSrc,
                  int H, int W, int cinShift, int taps, int Ktot, int relu) {
  __shared__ alignas(16) u16 lA[2][128 * 32];   // A tile [m][k]
  __shared__ alignas(16) u16 lB[2][128 * 32];   // B tile [n][k]

  const int tid   = threadIdx.x;
  const int mBase = blockIdx.x * 128;
  const int nBase = blockIdx.y * 128;
  const int lane  = tid & 31;
  const int wv    = tid >> 5;
  const int waveM = wv & 3;        // 4 M-subtiles of 32
  const int waveN = wv >> 2;       // 2 N-subtiles of 64
  const int HW    = H * W;
  const int Cin   = 1 << cinShift;
  const int cinMask = Cin - 1;
  const int nc    = Ktot >> 5;     // number of 32-wide K chunks

  // per-thread A-tile load coordinates (thread -> one 16-element half-row)
  const int aRow  = tid >> 1;          // 0..127
  const int aHalf = tid & 1;           // cols [aHalf*16, +16)
  const int mg    = mBase + aRow;
  const int bImg  = mg / HW;
  const int rem   = mg - bImg * HW;
  const int ay    = rem / W;
  const int ax    = rem - ay * W;

  const int h   = lane >> 4;
  const int l15 = lane & 15;

  v8f zero = {0.f, 0.f, 0.f, 0.f, 0.f, 0.f, 0.f, 0.f};
  v8f acc[2][4];
#pragma unroll
  for (int mi = 0; mi < 2; ++mi)
#pragma unroll
    for (int ni = 0; ni < 4; ++ni) acc[mi][ni] = zero;

  // per-thread A source pointer for a chunk (zeroSrc when padded)
  auto aSrc = [&](int chunk) -> const u16* {
    int k0 = chunk << 5;
    int t = k0 >> cinShift;
    int ci0 = k0 & cinMask;
    int dy = 0, dx = 0;
    if (taps == 9) { dy = t / 3 - 1; dx = t % 3 - 1; }
    int ys = ay + dy, xs = ax + dx;
    bool ok = (ys >= 0) && (ys < H) && (xs >= 0) && (xs < W);
    return ok ? act + ((size_t)(bImg * HW + ys * W + xs)) * Cin + ci0 + aHalf * 16
              : zeroSrc;
  };

#if USE_ASYNC
  // ---- A chunk: per-lane async DMA global -> LDS (no VGPR payload) ----
  auto asyncA = [&](int chunk, u16* buf) {
    const u16* src = aSrc(chunk);
    gvi4p g = (gvi4p)(uintptr_t)src;
    lvi4p l = (lvi4p)(uintptr_t)(u32)(size_t)(buf + aRow * 32 + aHalf * 16);
    __builtin_amdgcn_global_load_async_to_lds_b128(g, l, 0, 0);
    __builtin_amdgcn_global_load_async_to_lds_b128(g, l, 16, 0);
  };
#else
  // ---- fallback: register-pipelined sync staging ----
  auto ldA = [&](int chunk, uint4& v0, uint4& v1) {
    const uint4* src = (const uint4*)aSrc(chunk);
    v0 = src[0]; v1 = src[1];
  };
  auto stA = [&](u16* buf, const uint4& v0, const uint4& v1) {
    uint4* dst = (uint4*)(buf + aRow * 32 + aHalf * 16);
    dst[0] = v0; dst[1] = v1;
  };
#endif

#if USE_TDM
  // ---- B chunk DMA via Tensor Data Mover (issued once per workgroup) ----
  // D# (CDNA5 ISA): 2D tile, X = 32 k-elements, Y = 128 n-rows,
  // row stride = Ktot elements, data_size = 2 bytes, linear into LDS.
  auto tdmB = [&](int chunk, const u16* lbuf) {
    unsigned long long ga =
        (unsigned long long)(size_t)(wT + (size_t)nBase * Ktot + (chunk << 5));
    u32 laddr = (u32)(size_t)(const void*)lbuf;   // low 32 bits = LDS offset
    u32x4 g0;
    g0[0] = 1u;                                    // count=1, user mode
    g0[1] = laddr;                                 // lds_addr
    g0[2] = (u32)ga;                               // global_addr[31:0]
    g0[3] = ((u32)(ga >> 32) & 0x1FFFFFFu) | 0x80000000u; // addr[56:32] | type=2
    i32x8 g1;
    g1[0] = 1 << 16;          // data_size = 2 bytes, no multicast/pad/iterate
    g1[1] = 32 << 16;         // tensor_dim0 = 32
    g1[2] = 128 << 16;        // tensor_dim1 = 128
    g1[3] = 32 << 16;         // tile_dim0 = 32
    g1[4] = 128;              // tile_dim1 = 128, tile_dim2 = 0
    g1[5] = Ktot;             // tensor_dim0_stride[31:0]
    g1[6] = 0;
    g1[7] = 0;
    i32x4 z4; z4[0] = 0; z4[1] = 0; z4[2] = 0; z4[3] = 0;
    i32x8 z8; z8[0] = 0; z8[1] = 0; z8[2] = 0; z8[3] = 0;
    z8[4] = 0; z8[5] = 0; z8[6] = 0; z8[7] = 0;
    __builtin_amdgcn_tensor_load_to_lds(g0, g1, z4, z4, z8, 0);
  };
#else
  const int bRow  = tid >> 1;          // fallback sync B staging, mirrors A
  const int bHalf = tid & 1;
  auto ldB = [&](int chunk, uint4& v0, uint4& v1) {
    const uint4* src = (const uint4*)(wT +
        (size_t)(nBase + bRow) * Ktot + (chunk << 5) + bHalf * 16);
    v0 = src[0]; v1 = src[1];
  };
  auto stB = [&](u16* buf, const uint4& v0, const uint4& v1) {
    uint4* dst = (uint4*)(buf + bRow * 32 + bHalf * 16);
    dst[0] = v0; dst[1] = v1;
  };
#endif

  // ---- fragment gather + 8 WMMAs per chunk ----
  auto compute = [&](const u16* A, const u16* Bs) {
    Frag aF[2], bF[4];
#pragma unroll
    for (int mi = 0; mi < 2; ++mi) {
      const u16* p = A + (waveM * 32 + mi * 16 + l15) * 32 + h * 8;
      aF[mi].q[0] = *(const uint4*)p;          // K = h*8 .. +7
      aF[mi].q[1] = *(const uint4*)(p + 16);   // K = 16+h*8 .. +7
    }
#pragma unroll
    for (int ni = 0; ni < 4; ++ni) {
      const u16* p = Bs + (waveN * 64 + ni * 16 + l15) * 32 + h * 8;
      bF[ni].q[0] = *(const uint4*)p;          // K = h*8 .. +7
      bF[ni].q[1] = *(const uint4*)(p + 16);   // K = 16+h*8 .. +7
    }
#pragma unroll
    for (int mi = 0; mi < 2; ++mi)
#pragma unroll
      for (int ni = 0; ni < 4; ++ni)
        acc[mi][ni] = __builtin_amdgcn_wmma_f32_16x16x32_bf16(
            false, aF[mi].v, false, bF[ni].v, (short)0, acc[mi][ni], false, false);
  };

  // ---- prologue: stage chunk 0 into buffer 0 ----
  {
#if USE_ASYNC
    asyncA(0, lA[0]);
#else
    uint4 a0, a1;
    ldA(0, a0, a1);
    stA(lA[0], a0, a1);
#endif
#if USE_TDM
    if (wv == 0) tdmB(0, lB[0]);
#else
    uint4 b0, b1;
    ldB(0, b0, b1);
    stB(lB[0], b0, b1);
#endif
#if USE_ASYNC
    __builtin_amdgcn_s_wait_asynccnt(0);
#endif
#if USE_TDM
    if (wv == 0) __builtin_amdgcn_s_wait_tensorcnt(0);
#endif
    __syncthreads();
  }

  // ---- main pipelined loop ----
  for (int i = 0; i < nc; ++i) {
    const int cur = i & 1, nxt = cur ^ 1;
    const bool more = (i + 1 < nc);
#if USE_ASYNC
    if (more) {
      asyncA(i + 1, lA[nxt]);             // async DMA overlaps WMMAs below
#if USE_TDM
      if (wv == 0) tdmB(i + 1, lB[nxt]);  // TDM DMA overlaps WMMAs below
#endif
    }
    compute(lA[cur], lB[cur]);
    if (more) {
      __builtin_amdgcn_s_wait_asynccnt(0);            // own A loads landed
#if USE_TDM
      if (wv == 0) __builtin_amdgcn_s_wait_tensorcnt(0); // B DMA landed
#endif
      __syncthreads();                                 // publish both
    }
#else  // !USE_ASYNC: register-pipelined fallback
    uint4 a0, a1;
#if !USE_TDM
    uint4 b0, b1;
#endif
    if (more) {
      ldA(i + 1, a0, a1);
#if USE_TDM
      if (wv == 0) tdmB(i + 1, lB[nxt]);
#else
      ldB(i + 1, b0, b1);
#endif
    }
    compute(lA[cur], lB[cur]);
    if (more) {
      stA(lA[nxt], a0, a1);
#if USE_TDM
      __syncthreads();
      if (wv == 0) __builtin_amdgcn_s_wait_tensorcnt(0);
      __syncthreads();
#else
      stB(lB[nxt], b0, b1);
      __syncthreads();
#endif
    }
#endif
  }

  // ---- epilogue: per-channel scale/shift (+ReLU), store bf16 ----
#pragma unroll
  for (int mi = 0; mi < 2; ++mi)
#pragma unroll
    for (int ni = 0; ni < 4; ++ni) {
      int n = nBase + waveN * 64 + ni * 16 + l15;
      float sc = scale[n], sh = shift[n];
      int m0 = mBase + waveM * 32 + mi * 16 + h * 8;
#pragma unroll
      for (int r = 0; r < 8; ++r) {
        float v = acc[mi][ni][r] * sc + sh;
        if (relu) v = v > 0.f ? v : 0.f;
        out[(size_t)(m0 + r) * 256 + n] = f2bf(v);
      }
    }
}

// ---------------------------------------------------------------------------
// 5) 2x2 maxpool: [B*196,256] -> [B*49,256] (ReLU already applied upstream)
// ---------------------------------------------------------------------------
__global__ void maxpool_k(const u16* __restrict__ in, u16* __restrict__ out) {
  int i = blockIdx.x * 256 + threadIdx.x;
  if (i >= M7 * C256) return;
  int c = i & 255;
  int m = i >> 8;
  int b = m / 49, rem = m % 49, y = rem / 7, x = rem % 7;
  int r0 = b * 196 + (2 * y) * 14 + 2 * x;
  float v0 = bf2f(in[(size_t)r0 * 256 + c]);
  float v1 = bf2f(in[(size_t)(r0 + 1) * 256 + c]);
  float v2 = bf2f(in[(size_t)(r0 + 14) * 256 + c]);
  float v3 = bf2f(in[(size_t)(r0 + 15) * 256 + c]);
  out[i] = f2bf(fmaxf(fmaxf(v0, v1), fmaxf(v2, v3)));
}

// ---------------------------------------------------------------------------
// 6) bbox head: 3x3 conv 256->36, write [B,441,4] decode-layout raw preds
// ---------------------------------------------------------------------------
__global__ void bbox_head(const u16* __restrict__ act, const float* __restrict__ w,
                          const float* __restrict__ bias, float* __restrict__ bboxPred) {
  int i = blockIdx.x * 256 + threadIdx.x;
  if (i >= M7 * 36) return;
  int o = i % 36, m = i / 36;
  int b = m / 49, rem = m % 49, y = rem / 7, x = rem % 7;
  float acc = bias[o];
  for (int t = 0; t < 9; ++t) {
    int ys = y + t / 3 - 1, xs = x + t % 3 - 1;
    if ((unsigned)ys >= 7u || (unsigned)xs >= 7u) continue;
    const u16* ap = act + (size_t)(b * 49 + ys * 7 + xs) * 256;
    const float* wp = w + (size_t)o * K33 + (t / 3) * 3 + (t % 3);
    for (int ci = 0; ci < 256; ++ci) acc += bf2f(ap[ci]) * wp[(size_t)ci * 9];
  }
  int j = o >> 2, coord = o & 3;
  bboxPred[((size_t)b * NANCH + (y * 7 + x) * 9 + j) * 4 + coord] = acc;
}

// ---------------------------------------------------------------------------
// 7) conf head: 3x3 conv 256->9, sigmoid(raw*scale+bias), [B,441]
// ---------------------------------------------------------------------------
__global__ void conf_head(const u16* __restrict__ act, const float* __restrict__ w,
                          const float* __restrict__ bias, const float* __restrict__ cScale,
                          const float* __restrict__ cBias, float* __restrict__ confPred) {
  int i = blockIdx.x * 256 + threadIdx.x;
  if (i >= M7 * 9) return;
  int o = i % 9, m = i / 9;
  int b = m / 49, rem = m % 49, y = rem / 7, x = rem % 7;
  float acc = bias[o];
  for (int t = 0; t < 9; ++t) {
    int ys = y + t / 3 - 1, xs = x + t % 3 - 1;
    if ((unsigned)ys >= 7u || (unsigned)xs >= 7u) continue;
    const u16* ap = act + (size_t)(b * 49 + ys * 7 + xs) * 256;
    const float* wp = w + (size_t)o * K33 + (t / 3) * 3 + (t % 3);
    for (int ci = 0; ci < 256; ++ci) acc += bf2f(ap[ci]) * wp[(size_t)ci * 9];
  }
  float z = acc * cScale[0] + cBias[0];
  confPred[(size_t)b * NANCH + (y * 7 + x) * 9 + o] = 1.0f / (1.0f + expf(-z));
}

// ---------------------------------------------------------------------------
// 8) decode + clip + sort + greedy NMS + top-100, one workgroup per image
// ---------------------------------------------------------------------------
__device__ __forceinline__ float iou_fn(const float* a, const float* b) {
  float areaA = (a[2] - a[0]) * (a[3] - a[1]);
  float areaB = (b[2] - b[0]) * (b[3] - b[1]);
  float ix0 = fmaxf(a[0], b[0]), iy0 = fmaxf(a[1], b[1]);
  float ix1 = fminf(a[2], b[2]), iy1 = fminf(a[3], b[3]);
  float iw = fmaxf(ix1 - ix0, 0.f), ih = fmaxf(iy1 - iy0, 0.f);
  float inter = iw * ih;
  return inter / (areaA + areaB - inter + 1e-9f);
}

__global__ __launch_bounds__(512)
void nms_kernel(const float* __restrict__ bboxPred, const float* __restrict__ confPred,
                float* __restrict__ outBoxes, float* __restrict__ outScores,
                float* __restrict__ outValid) {
  __shared__ float key[NMSN];
  __shared__ int   sidx[NMSN];
  __shared__ float bx[NANCH][4];
  __shared__ float ssr[NANCH];
  __shared__ float sb[NANCH][4];
  __shared__ float ssc[NANCH];
  __shared__ int   keep[NANCH];
  __shared__ int   pos[NANCH];
  __shared__ int   nKeepSh;

  int b = blockIdx.x, tid = threadIdx.x;

  if (tid < NANCH) {
    int cell = tid / 9, s9 = tid % 9;
    int y = cell / 7, x = cell % 7;
    float cx = (x + 0.5f) / 7.0f, cy = (y + 0.5f) / 7.0f;
    const float scl[3] = {0.4f, 0.8f, 1.6f};
    const float rat[3] = {0.5f, 1.0f, 1.5f};
    float s = scl[s9 / 3], r = rat[s9 % 3];
    float sq = sqrtf(r);
    float aw = s * sq, ah = s / sq;
    const float* bp = bboxPred + ((size_t)b * NANCH + tid) * 4;
    float pcx = bp[0] * aw + cx, pcy = bp[1] * ah + cy;
    float pw = expf(bp[2]) * aw, ph = expf(bp[3]) * ah;
    float x0 = fminf(fmaxf(pcx - pw * 0.5f, 0.f), 1.f);
    float y0 = fminf(fmaxf(pcy - ph * 0.5f, 0.f), 1.f);
    float x1 = fminf(fmaxf(pcx + pw * 0.5f, 0.f), 1.f);
    float y1 = fminf(fmaxf(pcy + ph * 0.5f, 0.f), 1.f);
    bx[tid][0] = x0; bx[tid][1] = y0; bx[tid][2] = x1; bx[tid][3] = y1;
    float sc = confPred[(size_t)b * NANCH + tid];
    ssr[tid] = sc;
    key[tid] = (sc > 0.5f) ? sc : -1.0f;
  } else {
    key[tid] = -2.0f;                // padding sinks below invalid entries
  }
  sidx[tid] = tid;
  __syncthreads();

  // bitonic sort, descending by key
  for (int k = 2; k <= NMSN; k <<= 1) {
    for (int j = k >> 1; j > 0; j >>= 1) {
      int ixj = tid ^ j;
      if (ixj > tid) {
        bool up = ((tid & k) == 0);
        float a = key[tid], bb = key[ixj];
        bool sw = up ? (a < bb) : (a > bb);
        if (sw) {
          key[tid] = bb; key[ixj] = a;
          int ti = sidx[tid]; sidx[tid] = sidx[ixj]; sidx[ixj] = ti;
        }
      }
      __syncthreads();
    }
  }

  if (tid < NANCH) {
    int s0 = sidx[tid];
    sb[tid][0] = bx[s0][0]; sb[tid][1] = bx[s0][1];
    sb[tid][2] = bx[s0][2]; sb[tid][3] = bx[s0][3];
    ssc[tid] = ssr[s0];
    keep[tid] = (key[tid] > 0.5f) ? 1 : 0;
  }
  __syncthreads();

  // greedy suppression, sequential over sorted order (matches fori_loop)
  for (int i = 0; i < NANCH; ++i) {
    if (keep[i] && tid < NANCH && tid > i && keep[tid]) {
      if (iou_fn(sb[i], sb[tid]) > 0.5f) keep[tid] = 0;
    }
    __syncthreads();
  }

  // cap at 100 kept, compact kept-first (stable)
  if (tid == 0) {
    int ck = 0;
    for (int i = 0; i < NANCH; ++i)
      if (keep[i]) { ck++; if (ck > 100) keep[i] = 0; }
    int nk = 0;
    for (int i = 0; i < NANCH; ++i) if (keep[i]) pos[i] = nk++;
    int nn = 0;
    for (int i = 0; i < NANCH; ++i) if (!keep[i]) { pos[i] = nk + nn; nn++; }
    nKeepSh = nk;
  }
  __syncthreads();

  if (tid < NANCH) {
    int p = pos[tid];
    if (p < 100) {
      float* ob = outBoxes + ((size_t)b * 100 + p) * 4;
      ob[0] = sb[tid][0]; ob[1] = sb[tid][1]; ob[2] = sb[tid][2]; ob[3] = sb[tid][3];
      outScores[(size_t)b * 100 + p] = ssc[tid];
      outValid[(size_t)b * 100 + p] = keep[tid] ? 1.0f : 0.0f;
    }
  }
  __syncthreads();
  if (tid == 0 && nKeepSh == 0) {
    float* ob = outBoxes + (size_t)b * 100 * 4;
    ob[0] = 0.3f; ob[1] = 0.3f; ob[2] = 0.7f; ob[3] = 0.7f;
    outScores[(size_t)b * 100] = 0.5f;
    outValid[(size_t)b * 100] = 1.0f;
  }
}

// ---------------------------------------------------------------------------
// host-side launcher
// ---------------------------------------------------------------------------
extern "C" void kernel_launch(void* const* d_in, const int* in_sizes, int n_in,
                              void* d_out, int out_size, void* d_ws, size_t ws_size,
                              hipStream_t stream) {
  // pytree (sorted-key) flattening of setup_inputs(): features first, then params.
  const float* features     = (const float*)d_in[0];
  const float* bbox_b       = (const float*)d_in[1];
  const float* bbox_w       = (const float*)d_in[2];
  const float* cls1_b       = (const float*)d_in[3];
  const float* cls1_w       = (const float*)d_in[4];
  const float* cls1_beta    = (const float*)d_in[5];
  const float* cls1_gamma   = (const float*)d_in[6];
  const float* cls1_mean    = (const float*)d_in[7];
  const float* cls1_var     = (const float*)d_in[8];
  const float* cls2_b       = (const float*)d_in[9];
  const float* cls2_w       = (const float*)d_in[10];
  const float* conf_bias    = (const float*)d_in[11];
  const float* conf_scaling = (const float*)d_in[12];
  const float* conv1_b      = (const float*)d_in[13];
  const float* conv1_w      = (const float*)d_in[14];
  const float* conv1_beta   = (const float*)d_in[15];
  const float* conv1_gamma  = (const float*)d_in[16];
  const float* conv1_mean   = (const float*)d_in[17];
  const float* conv1_var    = (const float*)d_in[18];
  const float* conv2_b      = (const float*)d_in[19];
  const float* conv2_w      = (const float*)d_in[20];
  const float* conv2_beta   = (const float*)d_in[21];
  const float* conv2_gamma  = (const float*)d_in[22];
  const float* conv2_mean   = (const float*)d_in[23];
  const float* conv2_var    = (const float*)d_in[24];
  const float* lat_b        = (const float*)d_in[25];
  const float* lat_w        = (const float*)d_in[26];
  const float* lat_beta     = (const float*)d_in[27];
  const float* lat_gamma    = (const float*)d_in[28];
  const float* lat_mean     = (const float*)d_in[29];
  const float* lat_var      = (const float*)d_in[30];
  const float* pool_beta    = (const float*)d_in[31];
  const float* pool_gamma   = (const float*)d_in[32];
  const float* pool_mean    = (const float*)d_in[33];
  const float* pool_var     = (const float*)d_in[34];
  const float* pool_b       = (const float*)d_in[35];
  const float* pool_w       = (const float*)d_in[36];
  const float* smooth_b     = (const float*)d_in[37];
  const float* smooth_w     = (const float*)d_in[38];
  const float* smooth_beta  = (const float*)d_in[39];
  const float* smooth_gamma = (const float*)d_in[40];
  const float* smooth_mean  = (const float*)d_in[41];
  const float* smooth_var   = (const float*)d_in[42];

  char* ws = (char*)d_ws;
  size_t off = 0;
  auto take = [&](size_t bytes) -> void* {
    void* p = ws + off;
    off = (off + bytes + 255) & ~(size_t)255;
    return p;
  };

  u16* featsBF = (u16*)take((size_t)M14 * C512 * 2);
  u16* actA    = (u16*)take((size_t)M14 * C256 * 2);
  u16* actB    = (u16*)take((size_t)M14 * C256 * 2);
  u16* p0      = (u16*)take((size_t)M7 * C256 * 2);
  u16* p1      = (u16*)take((size_t)M7 * C256 * 2);
  u16* wtLat   = (u16*)take((size_t)C256 * C512 * 2);
  u16* wtSm    = (u16*)take((size_t)C256 * K33 * 2);
  u16* wtPool  = (u16*)take((size_t)C256 * K33 * 2);
  u16* wtC1    = (u16*)take((size_t)C256 * K33 * 2);
  u16* wtC2    = (u16*)take((size_t)C256 * K33 * 2);
  u16* wtCls1  = (u16*)take((size_t)C256 * K33 * 2);
  u16* zbuf    = (u16*)take(256);      // zero source for padded async lanes
  float* scLat  = (float*)take(256 * 4); float* shLat  = (float*)take(256 * 4);
  float* scSm   = (float*)take(256 * 4); float* shSm   = (float*)take(256 * 4);
  float* scPool = (float*)take(256 * 4); float* shPool = (float*)take(256 * 4);
  float* scC1   = (float*)take(256 * 4); float* shC1   = (float*)take(256 * 4);
  float* scC2   = (float*)take(256 * 4); float* shC2   = (float*)take(256 * 4);
  float* scCl   = (float*)take(256 * 4); float* shCl   = (float*)take(256 * 4);
  float* bboxPred = (float*)take((size_t)BATCH * NANCH * 4 * 4);
  float* confPred = (float*)take((size_t)BATCH * NANCH * 4);

  // --- preprocessing ---
  zero_fill<<<1, 128, 0, stream>>>(zbuf);
  {
    size_t N = (size_t)M14 * C512;
    cvt_features<<<(unsigned)((N + 255) / 256), 256, 0, stream>>>(features, featsBF);
  }
  pack_wT<<<(C256 * C512 + 255) / 256, 256, 0, stream>>>(lat_w, wtLat, 256, 512, 1);
  pack_wT<<<(C256 * K33 + 255) / 256, 256, 0, stream>>>(smooth_w, wtSm, 256, 256, 3);
  pack_wT<<<(C256 * K33 + 255) / 256, 256, 0, stream>>>(pool_w, wtPool, 256, 256, 3);
  pack_wT<<<(C256 * K33 + 255) / 256, 256, 0, stream>>>(conv1_w, wtC1, 256, 256, 3);
  pack_wT<<<(C256 * K33 + 255) / 256, 256, 0, stream>>>(conv2_w, wtC2, 256, 256, 3);
  pack_wT<<<(C256 * K33 + 255) / 256, 256, 0, stream>>>(cls1_w, wtCls1, 256, 256, 3);
  make_ss<<<1, 256, 0, stream>>>(lat_b,    lat_gamma,    lat_beta,    lat_mean,    lat_var,    scLat,  shLat);
  make_ss<<<1, 256, 0, stream>>>(smooth_b, smooth_gamma, smooth_beta, smooth_mean, smooth_var, scSm,   shSm);
  make_ss<<<1, 256, 0, stream>>>(pool_b,   pool_gamma,   pool_beta,   pool_mean,   pool_var,   scPool, shPool);
  make_ss<<<1, 256, 0, stream>>>(conv1_b,  conv1_gamma,  conv1_beta,  conv1_mean,  conv1_var,  scC1,   shC1);
  make_ss<<<1, 256, 0, stream>>>(conv2_b,  conv2_gamma,  conv2_beta,  conv2_mean,  conv2_var,  scC2,   shC2);
  make_ss<<<1, 256, 0, stream>>>(cls1_b,   cls1_gamma,   cls1_beta,   cls1_mean,   cls1_var,   scCl,   shCl);

  // --- WMMA conv trunk (block tile 128x128, gridDim.y = 256/128 = 2) ---
  dim3 g14(M14 / 128, 2), g7(M7 / 128, 2);
  gemm_bn_wmma<<<g14, 256, 0, stream>>>(featsBF, wtLat,  scLat,  shLat,  actA, zbuf, 14, 14, 9, 1, 512,  0); // lateral
  gemm_bn_wmma<<<g14, 256, 0, stream>>>(actA,    wtSm,   scSm,   shSm,   actB, zbuf, 14, 14, 8, 9, K33,  0); // smooth
  gemm_bn_wmma<<<g14, 256, 0, stream>>>(actB,    wtPool, scPool, shPool, actA, zbuf, 14, 14, 8, 9, K33,  1); // pool_conv+relu
  maxpool_k<<<(M7 * C256 + 255) / 256, 256, 0, stream>>>(actA, p0);
  gemm_bn_wmma<<<g7, 256, 0, stream>>>(p0, wtC1,   scC1, shC1, p1, zbuf, 7, 7, 8, 9, K33, 1);                // conv1
  gemm_bn_wmma<<<g7, 256, 0, stream>>>(p1, wtC2,   scC2, shC2, p0, zbuf, 7, 7, 8, 9, K33, 1);                // conv2 -> p0
  gemm_bn_wmma<<<g7, 256, 0, stream>>>(p0, wtCls1, scCl, shCl, p1, zbuf, 7, 7, 8, 9, K33, 1);                // cls1  -> p1

  // --- heads ---
  bbox_head<<<(M7 * 36 + 255) / 256, 256, 0, stream>>>(p0, bbox_w, bbox_b, bboxPred);
  conf_head<<<(M7 * 9 + 255) / 256, 256, 0, stream>>>(p1, cls2_w, cls2_b, conf_scaling, conf_bias, confPred);

  // --- decode + NMS ---
  float* outBoxes  = (float*)d_out;                       // [128,100,4]
  float* outScores = outBoxes + (size_t)BATCH * 100 * 4;  // [128,100]
  float* outValid  = outScores + (size_t)BATCH * 100;     // [128,100]
  nms_kernel<<<BATCH, 512, 0, stream>>>(bboxPred, confPred, outBoxes, outScores, outValid);
}
